// MultiHeadAttention_64106681860515
// MI455X (gfx1250) — matrix operand order repaired and checked by reading, not compiled, not run on hardware
//
#include <hip/hip_runtime.h>

// ---------------------------------------------------------------------------
// Causal MHA forward for MI455X (gfx1250, wave32, WMMA bf16 path).
//   x[B,T,C] f32 -> bf16; Wq/Wk/Wv pre-transposed to [H,D,C] bf16;
//   q (pre-scaled by D^-0.5), k [B,H,T,D] bf16; v transposed [B,H,D,T] bf16;
//   flash-attention per 16-query wave tile (online softmax, 32-key steps);
//   concat-head output ao[B,T,C] bf16; final GEMM x Wp^T + bp -> f32 out.
// Dense GEMMs use 32-row per-wave tiles (2 A-fragments x 4 N-tiles = 8 WMMA
// per k-step) to amortize B-fragment loads: ~1.5 b128 loads per WMMA.
// All matrix math: v_wmma_f32_16x16x32_bf16.
// ---------------------------------------------------------------------------

using u16 = unsigned short;
typedef __attribute__((ext_vector_type(16))) __bf16 v16bf;
typedef __attribute__((ext_vector_type(8)))  __bf16 v8bf;
typedef __attribute__((ext_vector_type(8)))  float  v8f;

constexpr int B = 4, T = 2048, C = 1024, H = 16, D = 64;

__device__ __forceinline__ u16 f2bf(float f) {
  unsigned int u = __float_as_uint(f);
  u += 0x7FFFu + ((u >> 16) & 1u);   // round-to-nearest-even
  return (u16)(u >> 16);
}

__device__ __forceinline__ v16bf ld16(const u16* p) {
  return *reinterpret_cast<const v16bf*>(p);
}
__device__ __forceinline__ v8bf ld8(const u16* p) {
  return *reinterpret_cast<const v8bf*>(p);
}

// A-fragment (16x32 bf16): lane holds row M=lane%16; elements e:
//   e<8  -> k = k0 + half*16 + e        (16B chunk)
//   e>=8 -> k = k0 + half*16 + e        (16B chunk)
__device__ __forceinline__ v16bf afrag(const u16* row, int k0, int hf) {
  v8bf lo = ld8(row + k0 + hf * 16);
  v8bf hi = ld8(row + k0 + hf * 16 + 8);
  return __builtin_shufflevector(lo, hi, 0, 1, 2, 3, 4, 5, 6, 7,
                                 8, 9, 10, 11, 12, 13, 14, 15);
}

__device__ __forceinline__ v8f wmma_bf16(v16bf a, v16bf b, v8f c) {
  return __builtin_amdgcn_wmma_f32_16x16x32_bf16(false, a, false, b,
                                                 (short)0, c, false, false);
}

// ---------------------------------------------------------------------------
__global__ void k_cvt(const float* __restrict__ in, u16* __restrict__ out, int n) {
  int i = blockIdx.x * 256 + threadIdx.x;
  if (i < n) out[i] = f2bf(in[i]);
}

// W [H,C,D] f32 -> WT [H,D,C] bf16
__global__ void k_wT(const float* __restrict__ in, u16* __restrict__ out) {
  int i = blockIdx.x * 256 + threadIdx.x;
  if (i >= H * C * D) return;
  int h = i / (C * D);
  int r = i % (C * D);
  int c = r / D;
  int d = r % D;
  out[((size_t)h * D + d) * C + c] = f2bf(in[i]);
}

// ---------------------------------------------------------------------------
// q/k/v projection: [T,C] x [C,D] per (b,h). Block = 8 waves, wave tile 32x64,
// block tile 256x64. oscale folds D^-0.5 into q. vtrans stores [D,T] for V.
__global__ __launch_bounds__(256) void k_qkv(const u16* __restrict__ xb,
                                             const u16* __restrict__ WT,
                                             u16* __restrict__ out, int vtrans,
                                             float oscale) {
  int wave = threadIdx.x >> 5, lane = threadIdx.x & 31;
  int n = lane & 15, hf = lane >> 4;
  int bh = blockIdx.y, b = bh / H, h = bh % H;
  int t0 = blockIdx.x * 256 + wave * 32;

  const u16* xrow0 = xb + (size_t)(b * T + t0 + n) * C;
  const u16* xrow1 = xb + (size_t)(b * T + t0 + 16 + n) * C;
  const u16* w0    = WT + (size_t)h * D * C;

  v8f z = {};
  v8f acc0[4] = {z, z, z, z};
  v8f acc1[4] = {z, z, z, z};
  for (int c0 = 0; c0 < C; c0 += 32) {
    v16bf a0 = afrag(xrow0, c0, hf);
    v16bf a1 = afrag(xrow1, c0, hf);
#pragma unroll
    for (int j = 0; j < 4; j++) {
      v16bf bf = ld16(w0 + (size_t)(j * 16 + n) * C + c0 + hf * 16);
      acc0[j] = wmma_bf16(a0, bf, acc0[j]);
      acc1[j] = wmma_bf16(a1, bf, acc1[j]);
    }
  }
#pragma unroll
  for (int j = 0; j < 4; j++) {
#pragma unroll
    for (int r = 0; r < 8; r++) {
      int d = j * 16 + n;
      int ta = t0 + r + 8 * hf;
      int tb = ta + 16;
      u16 va = f2bf(acc0[j][r] * oscale);
      u16 vb = f2bf(acc1[j][r] * oscale);
      if (vtrans) {
        out[((size_t)bh * D + d) * T + ta] = va;
        out[((size_t)bh * D + d) * T + tb] = vb;
      } else {
        out[((size_t)bh * T + ta) * D + d] = va;
        out[((size_t)bh * T + tb) * D + d] = vb;
      }
    }
  }
}

// ---------------------------------------------------------------------------
// Flash attention: each wave owns 16 query rows; 32 keys per iteration.
// q arrives pre-scaled by D^-0.5.
__global__ __launch_bounds__(256) void k_attn(const u16* __restrict__ q,
                                              const u16* __restrict__ k,
                                              const u16* __restrict__ vT,
                                              u16* __restrict__ ao) {
  __shared__ u16 psh[8][16][32];  // per-wave P tile (16 queries x 32 keys)

  int wave = threadIdx.x >> 5, lane = threadIdx.x & 31;
  int n = lane & 15, hf = lane >> 4;
  int bh = blockIdx.y, b = bh / H, h = bh % H;
  int t0 = blockIdx.x * 128 + wave * 16;

  const u16* qb = q  + (size_t)bh * T * D;
  const u16* kb = k  + (size_t)bh * T * D;
  const u16* vb = vT + (size_t)bh * D * T;

  const u16* qrow = qb + (size_t)(t0 + n) * D;
  v16bf qa0 = afrag(qrow, 0, hf);
  v16bf qa1 = afrag(qrow, 32, hf);

  float rmax[8], rsum[8];
#pragma unroll
  for (int r = 0; r < 8; r++) { rmax[r] = -1e30f; rsum[r] = 0.0f; }
  v8f z = {};
  v8f o[4] = {z, z, z, z};

  for (int s0 = 0; s0 < t0 + 16; s0 += 32) {
    // ---- S = Q K^T for two 16-key tiles (K-dim = D = 64 -> 2 wmma each)
    v8f sc0 = z, sc1 = z;
    {
      const u16* krow = kb + (size_t)(s0 + n) * D;
      sc0 = wmma_bf16(qa0, ld16(krow + hf * 16), sc0);
      sc0 = wmma_bf16(qa1, ld16(krow + 32 + hf * 16), sc0);
    }
    {
      const u16* krow = kb + (size_t)(s0 + 16 + n) * D;
      sc1 = wmma_bf16(qa0, ld16(krow + hf * 16), sc1);
      sc1 = wmma_bf16(qa1, ld16(krow + 32 + hf * 16), sc1);
    }
    // ---- online softmax (row = r + 8*half, col = lane%16)
#pragma unroll
    for (int r = 0; r < 8; r++) {
      int t = t0 + r + 8 * hf;
      float x0 = sc0[r]; if (s0 + n > t)      x0 = -1e30f;
      float x1 = sc1[r]; if (s0 + 16 + n > t) x1 = -1e30f;
      float mx = fmaxf(x0, x1);
#pragma unroll
      for (int off = 1; off < 16; off <<= 1)
        mx = fmaxf(mx, __shfl_xor(mx, off, 16));
      float nm = fmaxf(rmax[r], mx);
      float corr = __expf(rmax[r] - nm);
      rmax[r] = nm;
      float p0 = __expf(x0 - nm), p1 = __expf(x1 - nm);
      float ps = p0 + p1;
#pragma unroll
      for (int off = 1; off < 16; off <<= 1)
        ps += __shfl_xor(ps, off, 16);
      rsum[r] = rsum[r] * corr + ps;
      o[0][r] *= corr; o[1][r] *= corr; o[2][r] *= corr; o[3][r] *= corr;
      int mrow = r + 8 * hf;
      psh[wave][mrow][n]      = f2bf(p0);
      psh[wave][mrow][16 + n] = f2bf(p1);
    }
    // ensure LDS stores land before re-reading in A-fragment layout
    asm volatile("s_wait_dscnt 0x0" ::: "memory");
    v16bf pa = afrag(&psh[wave][n][0], 0, hf);
    // ---- O += P V  (V supplied transposed: rows = d, contiguous in s)
#pragma unroll
    for (int j = 0; j < 4; j++) {
      const u16* vrow = vb + (size_t)(j * 16 + n) * T;
      o[j] = wmma_bf16(pa, ld16(vrow + s0 + hf * 16), o[j]);
    }
  }

#pragma unroll
  for (int r = 0; r < 8; r++) {
    float inv = 1.0f / rsum[r];
    int t = t0 + r + 8 * hf;
#pragma unroll
    for (int j = 0; j < 4; j++)
      ao[(size_t)(b * T + t) * C + h * D + j * 16 + n] = f2bf(o[j][r] * inv);
  }
}

// ---------------------------------------------------------------------------
// out = ao @ Wp^T + bp.  B operand B[k=c, n=c'] = Wp[c',c] -> row-major Wp
// rows are already contiguous along k. Wave tile 32x64, block tile 256x64.
__global__ __launch_bounds__(256) void k_proj(const u16* __restrict__ ao,
                                              const u16* __restrict__ wp,
                                              const float* __restrict__ bp,
                                              float* __restrict__ out) {
  int wave = threadIdx.x >> 5, lane = threadIdx.x & 31;
  int n = lane & 15, hf = lane >> 4;
  int row0 = blockIdx.x * 256 + wave * 32;
  int n0 = blockIdx.y * 64;

  const u16* arow0 = ao + (size_t)(row0 + n) * C;
  const u16* arow1 = ao + (size_t)(row0 + 16 + n) * C;
  v8f z = {};
  v8f acc0[4] = {z, z, z, z};
  v8f acc1[4] = {z, z, z, z};
  for (int c0 = 0; c0 < C; c0 += 32) {
    v16bf a0 = afrag(arow0, c0, hf);
    v16bf a1 = afrag(arow1, c0, hf);
#pragma unroll
    for (int j = 0; j < 4; j++) {
      v16bf bf = ld16(wp + (size_t)(n0 + j * 16 + n) * C + c0 + hf * 16);
      acc0[j] = wmma_bf16(a0, bf, acc0[j]);
      acc1[j] = wmma_bf16(a1, bf, acc1[j]);
    }
  }
#pragma unroll
  for (int j = 0; j < 4; j++) {
    int cc = n0 + j * 16 + n;
    float bias = bp[cc];
#pragma unroll
    for (int r = 0; r < 8; r++) {
      out[(size_t)(row0 + r + 8 * hf) * C + cc]      = acc0[j][r] + bias;
      out[(size_t)(row0 + 16 + r + 8 * hf) * C + cc] = acc1[j][r] + bias;
    }
  }
}

// ---------------------------------------------------------------------------
extern "C" void kernel_launch(void* const* d_in, const int* in_sizes, int n_in,
                              void* d_out, int out_size, void* d_ws, size_t ws_size,
                              hipStream_t stream) {
  const float* x  = (const float*)d_in[0];
  const float* Wq = (const float*)d_in[1];
  const float* Wk = (const float*)d_in[2];
  const float* Wv = (const float*)d_in[3];
  const float* Wp = (const float*)d_in[4];
  const float* bp = (const float*)d_in[5];
  float* out = (float*)d_out;

  unsigned char* w = (unsigned char*)d_ws;
  auto take = [&](size_t nbytes) {
    void* p = (void*)w;
    w += (nbytes + 255) & ~(size_t)255;
    return p;
  };
  u16* xb  = (u16*)take((size_t)B * T * C * 2);
  u16* WqT = (u16*)take((size_t)H * D * C * 2);
  u16* WkT = (u16*)take((size_t)H * D * C * 2);
  u16* WvT = (u16*)take((size_t)H * D * C * 2);
  u16* Wpb = (u16*)take((size_t)C * C * 2);
  u16* qb  = (u16*)take((size_t)B * H * T * D * 2);
  u16* kb  = (u16*)take((size_t)B * H * T * D * 2);
  u16* vTb = (u16*)take((size_t)B * H * D * T * 2);
  u16* aob = (u16*)take((size_t)B * T * C * 2);

  int nx = B * T * C;
  k_cvt<<<(nx + 255) / 256, 256, 0, stream>>>(x, xb, nx);
  int nw = H * C * D;
  k_wT<<<(nw + 255) / 256, 256, 0, stream>>>(Wq, WqT);
  k_wT<<<(nw + 255) / 256, 256, 0, stream>>>(Wk, WkT);
  k_wT<<<(nw + 255) / 256, 256, 0, stream>>>(Wv, WvT);
  int np = C * C;
  k_cvt<<<(np + 255) / 256, 256, 0, stream>>>(Wp, Wpb, np);

  dim3 gq(T / 256, B * H);
  k_qkv<<<gq, 256, 0, stream>>>(xb, WqT, qb, 0, 0.125f);  // D^-0.5 folded in
  k_qkv<<<gq, 256, 0, stream>>>(xb, WkT, kb, 0, 1.0f);
  k_qkv<<<gq, 256, 0, stream>>>(xb, WvT, vTb, 1, 1.0f);

  dim3 ga(T / 128, B * H);
  k_attn<<<ga, 256, 0, stream>>>(qb, kb, vTb, aob);

  dim3 gp((B * T) / 256, C / 64);
  k_proj<<<gp, 256, 0, stream>>>(aob, Wpb, bp, out);
}
